// InvNet_24489903522663
// MI455X (gfx1250) — matrix operand order repaired and checked by reading, not compiled, not run on hardware
//
#include <hip/hip_runtime.h>
#include <hip/hip_bf16.h>
#include <math.h>

typedef __attribute__((ext_vector_type(16))) __bf16       v16bf;
typedef __attribute__((ext_vector_type(8)))  float        v8f;
typedef __attribute__((ext_vector_type(8)))  unsigned int v8u;

#define Bn     128
#define Dn     2048
#define Cn     16522
#define LDL    16544             // padded logits row stride: 517 blocks * 32 cols
#define GBLK   517               // ceil(16522/32)
#define KSTEPS 64                // 2048/32
#define BETA_INV 20.0f
#define KNNk   6

// f32 pair -> packed bf16 dword (RNE). Prefer the HW pack-convert if declared.
static __device__ __forceinline__ unsigned int pk_bf16(float lo, float hi) {
#if __has_builtin(__builtin_amdgcn_cvt_pk_bf16_f32)
  auto p = __builtin_amdgcn_cvt_pk_bf16_f32(lo, hi);
  static_assert(sizeof(p) == 4, "cvt_pk_bf16 size");
  return __builtin_bit_cast(unsigned int, p);
#else
  unsigned int a = __builtin_bit_cast(unsigned int, lo);
  unsigned int b = __builtin_bit_cast(unsigned int, hi);
  a += 0x7FFFu + ((a >> 16) & 1u);
  b += 0x7FFFu + ((b >> 16) & 1u);
  return (a >> 16) | (b & 0xFFFF0000u);
#endif
}

// ---------------------------------------------------------------------------
// Kernel 1: pre-pack inputs (A) into WMMA 16x32 bf16 fragment order.
// A layout: lane L -> M = L%16, h = L/16; VGPR v: K = 16h+2v(+1).
// ---------------------------------------------------------------------------
__global__ void pack_a_kernel(const float* __restrict__ inp,
                              unsigned int* __restrict__ apack) {
  int blk   = blockIdx.x;         // mtile*64 + kstep
  int mtile = blk >> 6;
  int kstep = blk & 63;
  int lane  = threadIdx.x;        // 0..31
  int h = lane >> 4, r = lane & 15;
  const float* src = inp + (size_t)(mtile * 16 + r) * Dn;
  unsigned int dw[8];
#pragma unroll
  for (int v = 0; v < 8; ++v) {
    int k0 = kstep * 32 + 16 * h + 2 * v;
    dw[v] = pk_bf16(src[k0], src[k0 + 1]);
  }
  uint4* dst = (uint4*)apack + ((size_t)blk * 32 + lane) * 2;
  dst[0] = make_uint4(dw[0], dw[1], dw[2], dw[3]);
  dst[1] = make_uint4(dw[4], dw[5], dw[6], dw[7]);
}

// ---------------------------------------------------------------------------
// Kernel 2: logits = (inputs @ em^T)/beta. Block = 8 waves, 32 em rows (2
// N-tiles). em panel converted once per block into LDS in B-fragment order
// (lane -> N = lane%16; lane half h -> K = 16h..16h+15, dword v = pair).
// K processed in 2 chunks of 1024 so the LDS stage buffer is 64 KB.
// ---------------------------------------------------------------------------
__global__ void __launch_bounds__(256) gemm_logits_kernel(
    const float* __restrict__ em,
    const unsigned int* __restrict__ apack,
    float* __restrict__ logits) {
  __shared__ unsigned int lds[16384];   // [ks(32)][nt(2)][lane(32)][v(8)] dwords

  int t    = threadIdx.x;
  int wave = t >> 5;                    // mtile 0..7
  int lane = t & 31;
  int h = lane >> 4, r = lane & 15;

  // staging role: thread t handles em row (block*32 + n32), 128 k per chunk
  int n32 = t & 31;
  int tg  = t >> 5;                     // k-group 0..7
  int crow = blockIdx.x * 32 + n32;
  if (crow >= Cn) crow = Cn - 1;        // clamp tail loads (stores guarded)
  const float* emr = em + (size_t)crow * Dn;
  int lane_sb = n32 & 15;
  int nt_s    = n32 >> 4;

  const uint4* ap = (const uint4*)apack + ((size_t)(wave * KSTEPS) * 32 + lane) * 2;
  v8f acc0 = {}, acc1 = {};

  for (int chunk = 0; chunk < 2; ++chunk) {
    // ---- stage: 8 runs of 16 consecutive k (one h-half of one kstep) ----
#pragma unroll
    for (int run = 0; run < 8; ++run) {
      int kl = tg * 128 + run * 16;             // k offset within chunk
      int ks = kl >> 5;                         // kstep within chunk
      int hh = (kl >> 4) & 1;
      const float4* src = (const float4*)(emr + chunk * 1024 + kl);
      float4 f0 = src[0], f1 = src[1], f2 = src[2], f3 = src[3];
      if (chunk == 0 && run == 0)
        __builtin_prefetch(emr + 1024 + tg * 128, 0, 0);  // next chunk
      unsigned int dw[8];
      dw[0] = pk_bf16(f0.x, f0.y); dw[1] = pk_bf16(f0.z, f0.w);
      dw[2] = pk_bf16(f1.x, f1.y); dw[3] = pk_bf16(f1.z, f1.w);
      dw[4] = pk_bf16(f2.x, f2.y); dw[5] = pk_bf16(f2.z, f2.w);
      dw[6] = pk_bf16(f3.x, f3.y); dw[7] = pk_bf16(f3.z, f3.w);
      int lane_s = hh * 16 + lane_sb;
      uint4* dst = (uint4*)&lds[(((ks * 2 + nt_s) * 32) + lane_s) * 8];
      dst[0] = make_uint4(dw[0], dw[1], dw[2], dw[3]);
      dst[1] = make_uint4(dw[4], dw[5], dw[6], dw[7]);
    }
    __syncthreads();

    // ---- compute: 32 ksteps, 2 N-tiles per wave ----
    for (int ks = 0; ks < 32; ++ks) {
      uint4 a0 = ap[0], a1 = ap[1];
      ap += 64;                                 // 32 lanes * 2 uint4 per kstep
      v8u au;
      au[0] = a0.x; au[1] = a0.y; au[2] = a0.z; au[3] = a0.w;
      au[4] = a1.x; au[5] = a1.y; au[6] = a1.z; au[7] = a1.w;
      v16bf avf = __builtin_bit_cast(v16bf, au);

      const uint4* bp0 = (const uint4*)&lds[((ks * 2 + 0) * 32 + lane) * 8];
      const uint4* bp1 = (const uint4*)&lds[((ks * 2 + 1) * 32 + lane) * 8];
      uint4 b00 = bp0[0], b01 = bp0[1];
      uint4 b10 = bp1[0], b11 = bp1[1];
      v8u bu0, bu1;
      bu0[0] = b00.x; bu0[1] = b00.y; bu0[2] = b00.z; bu0[3] = b00.w;
      bu0[4] = b01.x; bu0[5] = b01.y; bu0[6] = b01.z; bu0[7] = b01.w;
      bu1[0] = b10.x; bu1[1] = b10.y; bu1[2] = b10.z; bu1[3] = b10.w;
      bu1[4] = b11.x; bu1[5] = b11.y; bu1[6] = b11.z; bu1[7] = b11.w;
      v16bf bv0 = __builtin_bit_cast(v16bf, bu0);
      v16bf bv1 = __builtin_bit_cast(v16bf, bu1);

      acc0 = __builtin_amdgcn_wmma_f32_16x16x32_bf16(
          false, avf, false, bv0, (short)0, acc0, false, false);
      acc1 = __builtin_amdgcn_wmma_f32_16x16x32_bf16(
          false, avf, false, bv1, (short)0, acc1, false, false);
    }
    __syncthreads();
  }

  // C/D layout: N = lane%16, M = vgpr + 8*(lane/16)
#pragma unroll
  for (int nt = 0; nt < 2; ++nt) {
    int cc = blockIdx.x * 32 + nt * 16 + r;
    if (cc < Cn) {
#pragma unroll
      for (int vr = 0; vr < 8; ++vr) {
        int m = wave * 16 + vr + 8 * h;
        float v = (nt == 0) ? acc0[vr] : acc1[vr];
        logits[(size_t)m * LDL + cc] = v * BETA_INV;
      }
    }
  }
}

// ---------------------------------------------------------------------------
// Kernel 3: per-row logsumexp + top-6 + masked-CE loss (deterministic tree
// reductions with fixed pairing).
// ---------------------------------------------------------------------------
__global__ void __launch_bounds__(256) row_stats_kernel(
    const float* __restrict__ logits,
    const int* __restrict__ label,
    float* __restrict__ rowloss) {
  __shared__ float ms[256], ss[256];
  __shared__ float cval[256 * KNNk];
  __shared__ int   cidx[256 * KNNk];

  int row = blockIdx.x, t = threadIdx.x;
  const float* lp = logits + (size_t)row * LDL;

  float M = -INFINITY, S = 0.0f;
  float bv[KNNk]; int bi[KNNk];
#pragma unroll
  for (int q = 0; q < KNNk; ++q) { bv[q] = -INFINITY; bi[q] = -1; }

  for (int j = t; j < Cn; j += 256) {
    float x = lp[j];
    if (x > M) { S = S * expf(M - x) + 1.0f; M = x; }
    else       { S += expf(x - M); }
    if (x > bv[KNNk - 1]) {                  // sorted insert, register swap-chain
      float nv = x; int nj = j;
#pragma unroll
      for (int u = 0; u < KNNk; ++u) {
        if (nv > bv[u]) {
          float tv = bv[u]; int tj = bi[u];
          bv[u] = nv; bi[u] = nj; nv = tv; nj = tj;
        }
      }
    }
  }
  ms[t] = M; ss[t] = S;
#pragma unroll
  for (int q = 0; q < KNNk; ++q) { cval[t * KNNk + q] = bv[q]; cidx[t * KNNk + q] = bi[q]; }
  __syncthreads();

  for (int off = 128; off >= 1; off >>= 1) {
    if (t < off) {
      float m2 = ms[t + off], s2 = ss[t + off];
      if (m2 > M) { S = S * expf(M - m2) + s2; M = m2; }
      else        { S += s2 * expf(m2 - M); }
      ms[t] = M; ss[t] = S;
#pragma unroll
      for (int q = 0; q < KNNk; ++q) {       // insert partner's sorted list
        float nv = cval[(t + off) * KNNk + q];
        int   nj = cidx[(t + off) * KNNk + q];
#pragma unroll
        for (int u = 0; u < KNNk; ++u) {
          if (nv > bv[u]) {
            float tv = bv[u]; int tj = bi[u];
            bv[u] = nv; bi[u] = nj; nv = tv; nj = tj;
          }
        }
      }
#pragma unroll
      for (int q = 0; q < KNNk; ++q) { cval[t * KNNk + q] = bv[q]; cidx[t * KNNk + q] = bi[q]; }
    }
    __syncthreads();
  }

  if (t == 0) {
    float lse = M + logf(S);
    int lab = label[row];
    float sum_top = 0.0f; bool lab_in = false;
#pragma unroll
    for (int q = 0; q < KNNk; ++q) {
      sum_top += (lse - bv[q]);
      if (bi[q] == lab) lab_in = true;
    }
    float lv = lp[lab];
    rowloss[row] = 2.0f * sum_top + 3.0f * (lse - lv)
                 - (lab_in ? 2.0f * (lse - lv) : 0.0f);
  }
}

__global__ void final_loss_kernel(const float* __restrict__ rowloss,
                                  float* __restrict__ out) {
  if (blockIdx.x == 0 && threadIdx.x == 0) {
    float s = 0.0f;
    for (int i = 0; i < Bn; ++i) s += rowloss[i];
    out[0] = s / (float)Bn;
  }
}

// ---------------------------------------------------------------------------
// Kernel 4: sequential EMA update, parallel over label chains.
// ---------------------------------------------------------------------------
__global__ void __launch_bounds__(256) ema_update_kernel(
    const float* __restrict__ inp,
    const int* __restrict__ label,
    const int* __restrict__ epoch_p,
    float* __restrict__ emo) {
  __shared__ float red[256];
  int b = blockIdx.x, t = threadIdx.x;
  int y = label[b];
  for (int j = 0; j < b; ++j)
    if (label[j] == y) return;           // not a chain head (uniform exit)

  float alpha = 0.01f * (float)epoch_p[0];
  float beta1 = 1.0f - alpha;
  float* row = emo + (size_t)y * Dn;

  float e[8];
#pragma unroll
  for (int u = 0; u < 8; ++u) e[u] = row[t + 256 * u];

  for (int s = b; s < Bn; ++s) {
    if (label[s] != y) continue;
    const float* x = inp + (size_t)s * Dn;
    float ssq = 0.0f;
#pragma unroll
    for (int u = 0; u < 8; ++u) {
      e[u] = alpha * e[u] + beta1 * x[t + 256 * u];
      ssq += e[u] * e[u];
    }
    red[t] = ssq; __syncthreads();
    for (int off = 128; off > 0; off >>= 1) {
      if (t < off) red[t] += red[t + off];
      __syncthreads();
    }
    float inv = 1.0f / sqrtf(red[0]);
    __syncthreads();
#pragma unroll
    for (int u = 0; u < 8; ++u) e[u] *= inv;
  }
#pragma unroll
  for (int u = 0; u < 8; ++u) row[t + 256 * u] = e[u];
}

// ---------------------------------------------------------------------------
extern "C" void kernel_launch(void* const* d_in, const int* in_sizes, int n_in,
                              void* d_out, int out_size, void* d_ws, size_t ws_size,
                              hipStream_t stream) {
  (void)in_sizes; (void)n_in; (void)out_size; (void)ws_size;
  const float* inp   = (const float*)d_in[0];
  const int*   label = (const int*)d_in[1];
  const float* em    = (const float*)d_in[2];
  const int*   epoch = (const int*)d_in[3];
  float* out = (float*)d_out;

  char* ws = (char*)d_ws;
  unsigned int* apack = (unsigned int*)ws;                       // 512 KB
  float* logits  = (float*)(ws + 512 * 1024);                    // 128*16544*4
  float* rowloss = (float*)(ws + 512 * 1024 + (size_t)Bn * LDL * 4);

  hipMemcpyAsync(out + 1, em, (size_t)Cn * Dn * sizeof(float),
                 hipMemcpyDeviceToDevice, stream);

  pack_a_kernel<<<8 * KSTEPS, 32, 0, stream>>>(inp, apack);
  gemm_logits_kernel<<<GBLK, 256, 0, stream>>>(em, apack, logits);
  row_stats_kernel<<<Bn, 256, 0, stream>>>(logits, label, rowloss);
  final_loss_kernel<<<1, 64, 0, stream>>>(rowloss, out);
  ema_update_kernel<<<Bn, 256, 0, stream>>>(inp, label, epoch, out + 1);
}
